// WaveNetModel_2413771620510
// MI455X (gfx1250) — matrix operand order repaired and checked by reading, not compiled, not run on hardware
//
#include <hip/hip_runtime.h>
#include <hip/hip_bf16.h>
#include <math.h>

// ---------------------------------------------------------------------------
// WaveNet forward for gfx1250 (MI455X).  Core GEMMs use V_WMMA_F32_16X16X32_BF16
// (bf16 operands, fp32 accumulate).  Weights are pre-packed into the exact
// per-lane A order used here (contiguous k = 16*half + e, as in B) so the hot loop issues pure
// 128-bit global loads; activations are staged through LDS in the B-fragment
// layout.  One fused kernel per dilated layer keeps HBM traffic to
// one read + one write of the activation tensor per layer.
// ---------------------------------------------------------------------------

typedef __attribute__((ext_vector_type(16))) __bf16    bf16x16;
typedef __attribute__((ext_vector_type(8)))  float     f32x8;
typedef __attribute__((ext_vector_type(4)))  unsigned  u32x4;

union Frag {
    bf16x16        v;
    u32x4          q[2];
    unsigned short u[16];
};

#define N_LAYERS  19
#define T0_LEN    6135        // 6144 - 10 + 1
#define T_FINAL   4601        // 6135 - (1023 + 511)

static __device__ __forceinline__ unsigned short f2bf(float f) {
    unsigned u = __builtin_bit_cast(unsigned, f);
    unsigned r = u + 0x7FFFu + ((u >> 16) & 1u);   // round-to-nearest-even
    return (unsigned short)(r >> 16);
}

static __device__ __forceinline__ f32x8 zero8() {
    f32x8 z = {0.f, 0.f, 0.f, 0.f, 0.f, 0.f, 0.f, 0.f};
    return z;
}

// A-fragment K index for element e within a 32-wide K tile (contiguous order,
// matching B): lanes 0-15 hold K {0-15}, lanes 16-31 hold K {16-31}.
static __device__ __forceinline__ int a_klocal(int e, int half) {
    return half * 16 + e;
}

// ---------------------------------------------------------------------------
// Weight packing: dilated conv W[256 x 256] (K = tap*128 + c) -> A fragments.
// frag element p = ((layer<<16) | mt<<12 | kt<<9 | lane<<4 | e)
// ---------------------------------------------------------------------------
__global__ void pack_dil_kernel(const float* __restrict__ dil_w,
                                unsigned short* __restrict__ out) {
    int p = blockIdx.x * 256 + threadIdx.x;
    if (p >= N_LAYERS * 65536) return;
    int e     = p & 15;
    int lane  = (p >> 4) & 31;
    int kt    = (p >> 9) & 7;
    int mt    = (p >> 12) & 15;
    int layer = p >> 16;
    int half = lane >> 4, idx = lane & 15;
    int m = mt * 16 + idx;
    int k = kt * 32 + a_klocal(e, half);
    int tap = k >> 7, c = k & 127;
    out[p] = f2bf(dil_w[((layer * 256 + m) * 128 + c) * 2 + tap]);
}

// skip(256 rows) + res(128 rows) stacked -> W2[384 x 128] A fragments
__global__ void pack_sr_kernel(const float* __restrict__ skip_w,
                               const float* __restrict__ res_w,
                               unsigned short* __restrict__ out) {
    int p = blockIdx.x * 256 + threadIdx.x;
    if (p >= N_LAYERS * 49152) return;
    int layer = p / 49152;
    int rem   = p - layer * 49152;
    int e     = rem & 15;
    int lane  = (rem >> 4) & 31;
    int kt    = (rem >> 9) & 3;
    int mt2   = rem >> 11;                   // 0..23
    int half = lane >> 4, idx = lane & 15;
    int m2 = mt2 * 16 + idx;
    int c  = kt * 32 + a_klocal(e, half);
    float v = (m2 < 256) ? skip_w[(layer * 256 + m2) * 128 + c]
                         : res_w[(layer * 128 + (m2 - 256)) * 128 + c];
    out[p] = f2bf(v);
}

// condL[layer][b][o] = dil_b + cond_b + cond_w @ cvec   (19*4*256)
__global__ void cond_kernel(const float* __restrict__ cond_w,
                            const float* __restrict__ cond_b,
                            const float* __restrict__ dil_b,
                            const float* __restrict__ condition,
                            float* __restrict__ condL) {
    int p = blockIdx.x * 256 + threadIdx.x;
    if (p >= N_LAYERS * 1024) return;
    int o = p & 255, b = (p >> 8) & 3, layer = p >> 10;
    float s = cond_b[layer * 256 + o] + dil_b[layer * 256 + o];
    for (int c = 0; c < 80; ++c)
        s += cond_w[(layer * 256 + o) * 80 + c] * condition[b * 80 + c];
    condL[p] = s;
}

// cend[b][o] = cond_end_b + cond_end_w @ cvec  (4*256)
__global__ void cend_kernel(const float* __restrict__ w,
                            const float* __restrict__ bb,
                            const float* __restrict__ condition,
                            float* __restrict__ cend) {
    int p = blockIdx.x * 256 + threadIdx.x;
    if (p >= 1024) return;
    int o = p & 255, b = p >> 8;
    float s = bb[o];
    for (int c = 0; c < 80; ++c) s += w[o * 80 + c] * condition[b * 80 + c];
    cend[p] = s;
}

__global__ void zero_kernel(float* __restrict__ p, int n) {
    int i = blockIdx.x * 256 + threadIdx.x;
    if (i < n) p[i] = 0.f;
}

// Start conv: K=10, 60 -> 128 channels.  ~1.9 GMAC, memory trivial.
__global__ void start_conv_kernel(const float* __restrict__ in,
                                  const float* __restrict__ w,
                                  const float* __restrict__ bias,
                                  float* __restrict__ x0) {
    int p = blockIdx.x * 256 + threadIdx.x;
    if (p >= 4 * 128 * T0_LEN) return;
    int t = p % T0_LEN;
    int o = (p / T0_LEN) & 127;
    int b = p / (T0_LEN * 128);
    float s = bias[o];
    for (int c = 0; c < 60; ++c) {
        const float* ip = in + (b * 60 + c) * 6144 + t;
        const float* wp = w + (o * 60 + c) * 10;
#pragma unroll
        for (int j = 0; j < 10; ++j) s += ip[j] * wp[j];
    }
    x0[(b * 128 + o) * T0_LEN + t] = s;
}

// ---------------------------------------------------------------------------
// Fused dilated layer.  Block = 128 threads (4 waves), 1 batch x 32 samples.
//   Stage 1: D1[256,32] = Wdil[256x256] * [x(t); x(t+d)]  (8 WMMA K-steps)
//   gating : h = tanh(filt) * sigmoid(gate)  (filt/gate pairs same wave)
//   Stage 2: [skip;res][384,32] = W2[384x128] * h          (4 WMMA K-steps)
//   skip accumulated trailing-aligned into fixed 4601-sample buffer.
// ---------------------------------------------------------------------------
__global__ __launch_bounds__(128) void layer_kernel(
    const float* __restrict__ xin, float* __restrict__ xout,
    float* __restrict__ skip,
    const unsigned short* __restrict__ wdil,
    const unsigned short* __restrict__ wsr,
    const float* __restrict__ condL,     // [4][256] this layer (biases folded)
    const float* __restrict__ skip_b,    // [256]
    const float* __restrict__ res_b,     // [128]
    int d, int Tin, int Tout) {
    const int b    = blockIdx.y;
    const int t0   = blockIdx.x * 32;
    const int tid  = threadIdx.x;
    const int wv   = tid >> 5;
    const int lane = tid & 31;
    const int half = lane >> 4;
    const int idx  = lane & 15;

    __shared__ alignas(16) unsigned short ldsX[32 * 264];  // [t][k], k=tap*128+c
    __shared__ alignas(16) unsigned short ldsH[32 * 136];  // [t][c]

    // ---- stage x tile (both taps) into LDS as bf16, zero-padded ----
    for (int li = tid; li < 8192; li += 128) {
        int t = li & 31, k = li >> 5;
        int tap = k >> 7, c = k & 127;
        float v = (t0 + t < Tout) ? xin[(b * 128 + c) * Tin + t0 + t + tap * d]
                                  : 0.f;
        ldsX[t * 264 + k] = f2bf(v);
    }
    __syncthreads();

    // ---- stage 1: dilated GEMM, wave owns M-tiles {wv, wv+4, wv+8, wv+12} ----
    f32x8 acc[2][4];
#pragma unroll
    for (int nt = 0; nt < 2; ++nt)
#pragma unroll
        for (int j = 0; j < 4; ++j) acc[nt][j] = zero8();

#pragma unroll
    for (int kt = 0; kt < 8; ++kt) {
        Frag afr[4];
#pragma unroll
        for (int j = 0; j < 4; ++j) {
            int mt = wv + 4 * j;
            const u32x4* ap = reinterpret_cast<const u32x4*>(
                wdil + ((mt * 8 + kt) * 512 + lane * 16));
            afr[j].q[0] = ap[0];
            afr[j].q[1] = ap[1];
        }
#pragma unroll
        for (int nt = 0; nt < 2; ++nt) {
            Frag bfr;  // B layout: lane half h -> K = kt*32 + h*16 + e, N = idx
            const u32x4* bp = reinterpret_cast<const u32x4*>(
                &ldsX[(nt * 16 + idx) * 264 + kt * 32 + half * 16]);
            bfr.q[0] = bp[0];
            bfr.q[1] = bp[1];
#pragma unroll
            for (int j = 0; j < 4; ++j)
                acc[nt][j] = __builtin_amdgcn_wmma_f32_16x16x32_bf16(
                    false, afr[j].v, false, bfr.v, (short)0, acc[nt][j], false,
                    false);
        }
    }

    // ---- gating: tile pairs (j, j+2) hold channels (c, c+128) ----
    const float* cl = condL + b * 256;
#pragma unroll
    for (int nt = 0; nt < 2; ++nt) {
#pragma unroll
        for (int jp = 0; jp < 2; ++jp) {
            int mtf = wv + 4 * jp;
#pragma unroll
            for (int r = 0; r < 8; ++r) {
                int cf = mtf * 16 + r + 8 * half;
                float f = acc[nt][jp][r] + cl[cf];
                float g = acc[nt][jp + 2][r] + cl[cf + 128];
                float h = tanhf(f) * (1.f / (1.f + expf(-g)));
                ldsH[(nt * 16 + idx) * 136 + cf] = f2bf(h);
            }
        }
    }
    __syncthreads();

    // ---- stage 2: [skip;res] = W2[384x128] * h ----
    f32x8 acc2[2][6];
#pragma unroll
    for (int nt = 0; nt < 2; ++nt)
#pragma unroll
        for (int j = 0; j < 6; ++j) acc2[nt][j] = zero8();

#pragma unroll
    for (int kt = 0; kt < 4; ++kt) {
        Frag afr[6];
#pragma unroll
        for (int j = 0; j < 6; ++j) {
            int mt2 = wv + 4 * j;
            const u32x4* ap = reinterpret_cast<const u32x4*>(
                wsr + ((mt2 * 4 + kt) * 512 + lane * 16));
            afr[j].q[0] = ap[0];
            afr[j].q[1] = ap[1];
        }
#pragma unroll
        for (int nt = 0; nt < 2; ++nt) {
            Frag bfr;
            const u32x4* bp = reinterpret_cast<const u32x4*>(
                &ldsH[(nt * 16 + idx) * 136 + kt * 32 + half * 16]);
            bfr.q[0] = bp[0];
            bfr.q[1] = bp[1];
#pragma unroll
            for (int j = 0; j < 6; ++j)
                acc2[nt][j] = __builtin_amdgcn_wmma_f32_16x16x32_bf16(
                    false, afr[j].v, false, bfr.v, (short)0, acc2[nt][j], false,
                    false);
        }
    }

    // ---- writeback: skip accumulate (trailing aligned) + residual update ----
    const int Toff = Tout - T_FINAL;
#pragma unroll
    for (int nt = 0; nt < 2; ++nt) {
        int t = t0 + nt * 16 + idx;
        bool tin = (t < Tout);
#pragma unroll
        for (int j = 0; j < 6; ++j) {
            int mt2 = wv + 4 * j;  // uniform per wave-tile
#pragma unroll
            for (int r = 0; r < 8; ++r) {
                int m2 = mt2 * 16 + r + 8 * half;
                float v = acc2[nt][j][r];
                if (mt2 < 16) {  // skip channels 0..255
                    int tsk = t - Toff;
                    if (tin && tsk >= 0) {
                        int gi = (b * 256 + m2) * T_FINAL + tsk;
                        skip[gi] += v + skip_b[m2];
                    }
                } else {  // res channels
                    int rc = m2 - 256;
                    if (tin)
                        xout[(b * 128 + rc) * Tout + t] =
                            v + res_b[rc] + xin[(b * 128 + rc) * Tin + t + d];
                }
            }
        }
    }
}

// End: out = sigmoid(end_w @ tanh(skip + cend) + end_b); tanh tiled via LDS.
__global__ __launch_bounds__(256) void final_kernel(
    const float* __restrict__ skip, const float* __restrict__ cend,
    const float* __restrict__ end_w, const float* __restrict__ end_b,
    float* __restrict__ out) {
    int b = blockIdx.y;
    int t0 = blockIdx.x * 32;
    int tid = threadIdx.x;
    __shared__ float ts[256][33];
    for (int li = tid; li < 256 * 32; li += 256) {
        int t = li & 31, c = li >> 5;
        int tg = t0 + t;
        float v = 0.f;
        if (tg < T_FINAL)
            v = tanhf(skip[(b * 256 + c) * T_FINAL + tg] + cend[b * 256 + c]);
        ts[c][t] = v;
    }
    __syncthreads();
    for (int li = tid; li < 240 * 32; li += 256) {
        int t = li & 31, o = li >> 5;
        float s = end_b[o];
        for (int c = 0; c < 256; ++c) s += end_w[o * 256 + c] * ts[c][t];
        int tg = t0 + t;
        if (tg < T_FINAL)
            out[(b * 240 + o) * T_FINAL + tg] = 1.f / (1.f + expf(-s));
    }
}

// ---------------------------------------------------------------------------
extern "C" void kernel_launch(void* const* d_in, const int* in_sizes, int n_in,
                              void* d_out, int out_size, void* d_ws,
                              size_t ws_size, hipStream_t stream) {
    (void)in_sizes; (void)n_in; (void)out_size; (void)ws_size;
    const float* input   = (const float*)d_in[0];
    const float* condv   = (const float*)d_in[1];
    const float* start_w = (const float*)d_in[2];
    const float* start_b = (const float*)d_in[3];
    const float* dil_w   = (const float*)d_in[4];
    const float* dil_b   = (const float*)d_in[5];
    const float* cond_w  = (const float*)d_in[6];
    const float* cond_b  = (const float*)d_in[7];
    const float* res_w   = (const float*)d_in[8];
    const float* res_b   = (const float*)d_in[9];
    const float* skip_w  = (const float*)d_in[10];
    const float* skip_b  = (const float*)d_in[11];
    const float* end_w   = (const float*)d_in[12];
    const float* end_b   = (const float*)d_in[13];
    const float* cend_w  = (const float*)d_in[14];
    const float* cend_b  = (const float*)d_in[15];
    float* out = (float*)d_out;

    // workspace carve-up (~48 MB)
    char* ws = (char*)d_ws;
    size_t off = 0;
    auto carve = [&](size_t bytes) -> char* {
        char* p = ws + off;
        off += (bytes + 255) & ~(size_t)255;
        return p;
    };
    float* xA    = (float*)carve((size_t)4 * 128 * T0_LEN * sizeof(float));
    float* xB    = (float*)carve((size_t)4 * 128 * T0_LEN * sizeof(float));
    float* skip  = (float*)carve((size_t)4 * 256 * T_FINAL * sizeof(float));
    float* condL = (float*)carve((size_t)N_LAYERS * 1024 * sizeof(float));
    float* cendb = (float*)carve((size_t)1024 * sizeof(float));
    unsigned short* pdil =
        (unsigned short*)carve((size_t)N_LAYERS * 65536 * sizeof(unsigned short));
    unsigned short* psr =
        (unsigned short*)carve((size_t)N_LAYERS * 49152 * sizeof(unsigned short));

    pack_dil_kernel<<<(N_LAYERS * 65536 + 255) / 256, 256, 0, stream>>>(dil_w,
                                                                        pdil);
    pack_sr_kernel<<<(N_LAYERS * 49152 + 255) / 256, 256, 0, stream>>>(
        skip_w, res_w, psr);
    cond_kernel<<<(N_LAYERS * 1024 + 255) / 256, 256, 0, stream>>>(
        cond_w, cond_b, dil_b, condv, condL);
    cend_kernel<<<4, 256, 0, stream>>>(cend_w, cend_b, condv, cendb);
    zero_kernel<<<(4 * 256 * T_FINAL + 255) / 256, 256, 0, stream>>>(
        skip, 4 * 256 * T_FINAL);
    start_conv_kernel<<<(4 * 128 * T0_LEN + 255) / 256, 256, 0, stream>>>(
        input, start_w, start_b, xA);

    int dils[N_LAYERS];
    {
        int n = 0;
        for (int blk = 0; blk < 2; ++blk) {
            int L = (blk < 1) ? 10 : 9;
            for (int i = 0; i < L; ++i) dils[n++] = 1 << i;
        }
    }

    int Tin = T0_LEN;
    float* cur = xA;
    float* nxt = xB;
    for (int i = 0; i < N_LAYERS; ++i) {
        int dd = dils[i];
        int Tout = Tin - dd;
        dim3 grid((Tout + 31) / 32, 4);
        layer_kernel<<<grid, 128, 0, stream>>>(
            cur, nxt, skip, pdil + (size_t)i * 65536, psr + (size_t)i * 49152,
            condL + i * 1024, skip_b + i * 256, res_b + i * 128, dd, Tin, Tout);
        float* tmp = cur; cur = nxt; nxt = tmp;
        Tin = Tout;
    }

    dim3 fgrid((T_FINAL + 31) / 32, 4);
    final_kernel<<<fgrid, 256, 0, stream>>>(skip, cendb, end_w, end_b, out);
}
